// MultiQueryAttention_5884105195698
// MI455X (gfx1250) — matrix operand order repaired and checked by reading, not compiled, not run on hardware
//
#include <hip/hip_runtime.h>

// Problem constants (B=2, S=2048, D=2048, H=16, HD=128)
#define BB 2
#define SS 2048
#define DD 2048
#define HH 16
#define HDD 128

typedef __attribute__((ext_vector_type(16))) __bf16 bf16x16;
typedef __attribute__((ext_vector_type(8))) float floatx8;

union BfOp {
    bf16x16 v;
    unsigned short us[16];
    uint4 q2[2];
};

__device__ __forceinline__ unsigned short f2bf(float f) {
    unsigned int u = __float_as_uint(f);
    return (unsigned short)((u + 0x7FFFu + ((u >> 16) & 1u)) >> 16);  // RTNE
}

// ---------------------------------------------------------------------------
// CDNA5 async global -> LDS copy (ASYNCcnt-tracked) helpers
// ---------------------------------------------------------------------------
__device__ __forceinline__ void async_copy_b128(const unsigned short* gptr,
                                                unsigned short* lptr) {
    // LDS aperture: addr[31:0] of the generic pointer is the LDS byte offset.
    unsigned lds = (unsigned)(uintptr_t)lptr;
    asm volatile("global_load_async_to_lds_b128 %0, %1, off"
                 :
                 : "v"(lds), "v"(gptr)
                 : "memory");
}
__device__ __forceinline__ void wait_async0() {
    asm volatile("s_wait_asynccnt 0" ::: "memory");
}

// ---------------------------------------------------------------------------
// fp32 -> bf16 elementwise conversion
// ---------------------------------------------------------------------------
__global__ __launch_bounds__(256) void f2bf_kernel(const float* __restrict__ in,
                                                   unsigned short* __restrict__ out,
                                                   int n) {
    int idx = blockIdx.x * 256 + threadIdx.x;
    if (idx < n) out[idx] = f2bf(in[idx]);
}

// ---------------------------------------------------------------------------
// C[M,N] = A[M,K] @ W[N,K]^T  (bf16 in, f32 out)
// Block: 256 thr = 8 waves (2x4); macrotile 128x128, k-stage 32, LDS double-buffered
// via async global->LDS copies. Each wave: 64x32 = 4x2 WMMA accumulators.
// ---------------------------------------------------------------------------
#define BMT 128
#define BNT 128
#define BKT 32
#define LDP 40  // padded LDS row length in bf16 elements (80 B, 16B-aligned)

__global__ __launch_bounds__(256) void gemm_bf16_lds(const unsigned short* __restrict__ A,
                                                     const unsigned short* __restrict__ W,
                                                     float* __restrict__ C,
                                                     int M, int N, int K) {
    __shared__ unsigned short sm[2 * 2 * BMT * LDP];  // 40 KB: [buf][A|W][128][40]
    const int tid = threadIdx.x;
    const int wave = tid >> 5, lane = tid & 31;
    const int h = lane >> 4, n = lane & 15;
    const int wave_m = wave >> 2;  // 0..1 -> 64 rows each
    const int wave_n = wave & 3;   // 0..3 -> 32 cols each
    const int bm = blockIdx.x, bn = blockIdx.y;

    // cooperative copy mapping: 2 threads per row, 32B (2 x b128) per thread
    const int crow = tid >> 1;
    const int cseg = (tid & 1) * 16;  // element offset within 32-elem k-stage

    unsigned short* As0 = sm;
    unsigned short* Ws0 = sm + BMT * LDP;
    const int bufstride = 2 * BMT * LDP;

    const unsigned short* ga_base = A + (size_t)(bm * BMT + crow) * K + cseg;
    const unsigned short* gw_base = W + (size_t)(bn * BNT + crow) * K + cseg;

#define ISSUE_STAGE(buf, kc)                                                   \
    do {                                                                       \
        unsigned short* la = As0 + (buf)*bufstride + crow * LDP + cseg;        \
        unsigned short* lw = Ws0 + (buf)*bufstride + crow * LDP + cseg;        \
        async_copy_b128(ga_base + (kc), la);                                   \
        async_copy_b128(ga_base + (kc) + 8, la + 8);                           \
        async_copy_b128(gw_base + (kc), lw);                                   \
        async_copy_b128(gw_base + (kc) + 8, lw + 8);                           \
    } while (0)

    floatx8 acc[4][2];
#pragma unroll
    for (int tm = 0; tm < 4; ++tm)
#pragma unroll
        for (int tn = 0; tn < 2; ++tn) acc[tm][tn] = floatx8{};

    const int nk = K / BKT;
    ISSUE_STAGE(0, 0);
    wait_async0();
    __syncthreads();

    for (int ks = 0; ks < nk; ++ks) {
        const int buf = ks & 1;
        if (ks + 1 < nk) ISSUE_STAGE(buf ^ 1, (ks + 1) * BKT);

        const unsigned short* Ab = As0 + buf * bufstride;
        const unsigned short* Wb = Ws0 + buf * bufstride;
        // B-fragments (hoisted, reused by all 4 row-tiles)
        BfOp bfr[2];
#pragma unroll
        for (int tn = 0; tn < 2; ++tn) {
            const unsigned short* p = Wb + (wave_n * 32 + tn * 16 + n) * LDP + h * 16;
            bfr[tn].q2[0] = *(const uint4*)p;
            bfr[tn].q2[1] = *(const uint4*)(p + 8);
        }
#pragma unroll
        for (int tm = 0; tm < 4; ++tm) {
            const unsigned short* p = Ab + (wave_m * 64 + tm * 16 + n) * LDP + h * 16;
            BfOp af;
            af.q2[0] = *(const uint4*)p;
            af.q2[1] = *(const uint4*)(p + 8);
#pragma unroll
            for (int tn = 0; tn < 2; ++tn)
                acc[tm][tn] = __builtin_amdgcn_wmma_f32_16x16x32_bf16(
                    false, af.v, false, bfr[tn].v, (short)0, acc[tm][tn], false, false);
        }
        wait_async0();
        __syncthreads();
    }
#undef ISSUE_STAGE

    // epilogue: C layout element (m = r + 8h, n = lane%16) in vgpr r
#pragma unroll
    for (int tm = 0; tm < 4; ++tm) {
#pragma unroll
        for (int tn = 0; tn < 2; ++tn) {
            float* crow_p = C +
                (size_t)(bm * BMT + wave_m * 64 + tm * 16 + 8 * h) * N +
                bn * BNT + wave_n * 32 + tn * 16 + n;
#pragma unroll
            for (int r = 0; r < 8; ++r) crow_p[(size_t)r * N] = acc[tm][tn][r];
        }
    }
}

// ---------------------------------------------------------------------------
// RoPE for Q: [B,S,H,HD] fp32 -> bf16, pairwise complex rotation
// ---------------------------------------------------------------------------
__global__ __launch_bounds__(256) void rope_q_kernel(const float* __restrict__ Qf,
                                                     const float* __restrict__ cs,
                                                     const float* __restrict__ sn,
                                                     unsigned short* __restrict__ Qb) {
    int idx = blockIdx.x * 256 + threadIdx.x;  // over B*S*H*64 pairs
    int i = idx & 63;
    int hh = (idx >> 6) & 15;
    int s = (idx >> 10) & (SS - 1);
    int b = idx >> 21;
    size_t base = ((size_t)(b * SS + s)) * DD + hh * HDD + 2 * i;
    float e = Qf[base], o = Qf[base + 1];
    float c = cs[s * 64 + i], si = sn[s * 64 + i];
    Qb[base] = f2bf(e * c - o * si);
    Qb[base + 1] = f2bf(e * si + o * c);
}

// RoPE for K: [B,S,HD] fp32 -> bf16
__global__ __launch_bounds__(256) void rope_k_kernel(const float* __restrict__ Kf,
                                                     const float* __restrict__ cs,
                                                     const float* __restrict__ sn,
                                                     unsigned short* __restrict__ Kb) {
    int idx = blockIdx.x * 256 + threadIdx.x;  // over B*S*64 pairs
    int i = idx & 63;
    int s = (idx >> 6) & (SS - 1);
    int b = idx >> 17;
    size_t base = ((size_t)(b * SS + s)) * HDD + 2 * i;
    float e = Kf[base], o = Kf[base + 1];
    float c = cs[s * 64 + i], si = sn[s * 64 + i];
    Kb[base] = f2bf(e * c - o * si);
    Kb[base + 1] = f2bf(e * si + o * c);
}

// V transpose: [B,S,HD] fp32 -> [B,HD,S] bf16 (PV B-operand loads become contiguous)
__global__ __launch_bounds__(256) void conv_vt_kernel(const float* __restrict__ Vf,
                                                      unsigned short* __restrict__ Vtb) {
    int idx = blockIdx.x * 256 + threadIdx.x;  // over B*S*HD
    int c = idx & (HDD - 1);
    int s = (idx >> 7) & (SS - 1);
    int b = idx >> 18;
    Vtb[((size_t)(b * HDD + c)) * SS + s] = f2bf(Vf[((size_t)(b * SS + s)) * HDD + c]);
}

// ---------------------------------------------------------------------------
// Flash MQA: one wave per (batch, head, 16-query tile); 32-key blocks with
// online softmax. Score + PV matmuls via bf16 WMMA.
// ---------------------------------------------------------------------------
__global__ __launch_bounds__(256) void mqa_attn_kernel(const unsigned short* __restrict__ Qb,
                                                       const unsigned short* __restrict__ Kb,
                                                       const unsigned short* __restrict__ Vtb,
                                                       unsigned short* __restrict__ Ob) {
    __shared__ float Pl[8 * 16 * 32];  // 2 KB per wave P-bounce buffer
    int wave = threadIdx.x >> 5, lane = threadIdx.x & 31;
    int h = lane >> 4, n = lane & 15;
    int gw = blockIdx.x * 8 + wave;
    int qt = gw & 127;          // S/16 = 128 query tiles
    int head = (gw >> 7) & 15;  // 16 heads
    int b = gw >> 11;           // batch
    int q0 = qt << 4;
    float* P = Pl + wave * 512;

    // Q tile as 4 A-operands (HD = 4 x 32)
    BfOp qa[4];
    const unsigned short* qrow = Qb + ((size_t)(b * SS + q0 + n)) * DD + head * HDD;
#pragma unroll
    for (int c = 0; c < 4; ++c) {
        qa[c].q2[0] = *(const uint4*)(qrow + c * 32 + h * 16);
        qa[c].q2[1] = *(const uint4*)(qrow + c * 32 + h * 16 + 8);
    }

    floatx8 acc[8];
#pragma unroll
    for (int t = 0; t < 8; ++t) acc[t] = floatx8{};
    float mrow[8], lrow[8];
#pragma unroll
    for (int r = 0; r < 8; ++r) { mrow[r] = -1e30f; lrow[r] = 0.f; }
    const float scale = 0.08838834764831845f;  // 1/sqrt(128)

    int nkb = (q0 + 47) >> 5;  // 32-key blocks touching the causal region
    for (int kbi = 0; kbi < nkb; ++kbi) {
        int kb = kbi << 5;
        floatx8 s[2];
#pragma unroll
        for (int t = 0; t < 2; ++t) {
            const unsigned short* krow =
                Kb + ((size_t)(b * SS + kb + t * 16 + n)) * HDD + h * 16;
            floatx8 sc = {};
#pragma unroll
            for (int c = 0; c < 4; ++c) {
                BfOp kv;
                kv.q2[0] = *(const uint4*)(krow + c * 32);
                kv.q2[1] = *(const uint4*)(krow + c * 32 + 8);
                sc = __builtin_amdgcn_wmma_f32_16x16x32_bf16(false, qa[c].v, false, kv.v,
                                                             (short)0, sc, false, false);
            }
            s[t] = sc;
        }
        float mx[8];
#pragma unroll
        for (int r = 0; r < 8; ++r) {
            int qpos = q0 + r + 8 * h;
            float v0 = (kb + n <= qpos) ? s[0][r] * scale : -1e30f;
            float v1 = (kb + 16 + n <= qpos) ? s[1][r] * scale : -1e30f;
            s[0][r] = v0;
            s[1][r] = v1;
            mx[r] = fmaxf(v0, v1);
        }
#pragma unroll
        for (int m = 1; m < 16; m <<= 1) {
#pragma unroll
            for (int r = 0; r < 8; ++r) mx[r] = fmaxf(mx[r], __shfl_xor(mx[r], m, 32));
        }
        float alpha[8], rs[8];
#pragma unroll
        for (int r = 0; r < 8; ++r) {
            float mnew = fmaxf(mrow[r], mx[r]);
            alpha[r] = __expf(mrow[r] - mnew);
            float p0 = __expf(s[0][r] - mnew);
            float p1 = __expf(s[1][r] - mnew);
            s[0][r] = p0;
            s[1][r] = p1;
            rs[r] = p0 + p1;
            mrow[r] = mnew;
        }
#pragma unroll
        for (int m = 1; m < 16; m <<= 1) {
#pragma unroll
            for (int r = 0; r < 8; ++r) rs[r] += __shfl_xor(rs[r], m, 32);
        }
#pragma unroll
        for (int r = 0; r < 8; ++r) lrow[r] = lrow[r] * alpha[r] + rs[r];
#pragma unroll
        for (int t = 0; t < 8; ++t) {
#pragma unroll
            for (int r = 0; r < 8; ++r) acc[t][r] *= alpha[r];
        }
        // P: C-layout -> LDS -> A-layout (cross-lane shuffle via LDS, wave-private)
#pragma unroll
        for (int r = 0; r < 8; ++r) {
            P[(r + 8 * h) * 32 + n] = s[0][r];
            P[(r + 8 * h) * 32 + 16 + n] = s[1][r];
        }
        BfOp pa;
        const float* prow = P + n * 32 + h * 16;
#pragma unroll
        for (int j = 0; j < 8; ++j) {
            pa.us[j] = f2bf(prow[j]);
            pa.us[8 + j] = f2bf(prow[8 + j]);
        }
#pragma unroll
        for (int t = 0; t < 8; ++t) {
            const unsigned short* vrow =
                Vtb + ((size_t)(b * HDD + t * 16 + n)) * SS + kb + h * 16;
            BfOp vv;
            vv.q2[0] = *(const uint4*)(vrow);
            vv.q2[1] = *(const uint4*)(vrow + 8);
            acc[t] = __builtin_amdgcn_wmma_f32_16x16x32_bf16(false, pa.v, false, vv.v,
                                                             (short)0, acc[t], false, false);
        }
    }
#pragma unroll
    for (int t = 0; t < 8; ++t) {
#pragma unroll
        for (int r = 0; r < 8; ++r) {
            float o = acc[t][r] / lrow[r];
            Ob[((size_t)(b * SS + q0 + r + 8 * h)) * DD + head * HDD + t * 16 + n] = f2bf(o);
        }
    }
}

// ---------------------------------------------------------------------------
// Host launch
// ---------------------------------------------------------------------------
extern "C" void kernel_launch(void* const* d_in, const int* in_sizes, int n_in,
                              void* d_out, int out_size, void* d_ws, size_t ws_size,
                              hipStream_t stream) {
    const float* x = (const float*)d_in[0];
    const float* Wq = (const float*)d_in[1];
    const float* Wk = (const float*)d_in[2];
    const float* Wv = (const float*)d_in[3];
    const float* Wo = (const float*)d_in[4];
    const float* fc = (const float*)d_in[5];
    const float* fs = (const float*)d_in[6];
    float* out = (float*)d_out;

    char* w = (char*)d_ws;
    const size_t MB = 1u << 20;
    unsigned short* xb  = (unsigned short*)(w + 0 * MB);   // 16 MB
    unsigned short* wqb = (unsigned short*)(w + 16 * MB);  // 8 MB
    unsigned short* wkb = (unsigned short*)(w + 24 * MB);  // 0.5 MB
    unsigned short* wvb = (unsigned short*)(w + 25 * MB);  // 0.5 MB
    unsigned short* wob = (unsigned short*)(w + 26 * MB);  // 8 MB
    float* Qf           = (float*)(w + 34 * MB);           // 32 MB
    float* Kf           = (float*)(w + 66 * MB);           // 2 MB
    float* Vf           = (float*)(w + 68 * MB);           // 2 MB
    unsigned short* Qb  = (unsigned short*)(w + 70 * MB);  // 16 MB
    unsigned short* Kb  = (unsigned short*)(w + 86 * MB);  // 1 MB
    unsigned short* Vtb = (unsigned short*)(w + 87 * MB);  // 1 MB
    unsigned short* Ob  = (unsigned short*)(w + 34 * MB);  // alias Qf (dead by then)

    const int M = BB * SS;  // 4096

    // bf16 conversions
    f2bf_kernel<<<(M * DD) / 256, 256, 0, stream>>>(x, xb, M * DD);
    f2bf_kernel<<<(DD * DD) / 256, 256, 0, stream>>>(Wq, wqb, DD * DD);
    f2bf_kernel<<<(HDD * DD) / 256, 256, 0, stream>>>(Wk, wkb, HDD * DD);
    f2bf_kernel<<<(HDD * DD) / 256, 256, 0, stream>>>(Wv, wvb, HDD * DD);
    f2bf_kernel<<<(DD * DD) / 256, 256, 0, stream>>>(Wo, wob, DD * DD);

    // projections (LDS-tiled async WMMA GEMMs)
    gemm_bf16_lds<<<dim3(M / BMT, DD / BNT), 256, 0, stream>>>(xb, wqb, Qf, M, DD, DD);
    gemm_bf16_lds<<<dim3(M / BMT, HDD / BNT), 256, 0, stream>>>(xb, wkb, Kf, M, HDD, DD);
    gemm_bf16_lds<<<dim3(M / BMT, HDD / BNT), 256, 0, stream>>>(xb, wvb, Vf, M, HDD, DD);

    // RoPE + layout conversions to bf16
    rope_q_kernel<<<(BB * SS * HH * 64) / 256, 256, 0, stream>>>(Qf, fc, fs, Qb);
    rope_k_kernel<<<(BB * SS * 64) / 256, 256, 0, stream>>>(Kf, fc, fs, Kb);
    conv_vt_kernel<<<(BB * SS * HDD) / 256, 256, 0, stream>>>(Vf, Vtb);

    // flash MQA (4096 waves: B*H*(S/16))
    mqa_attn_kernel<<<(BB * HH * (SS / 16)) / 8, 256, 0, stream>>>(Qb, Kb, Vtb, Ob);

    // output projection straight into d_out
    gemm_bf16_lds<<<dim3(M / BMT, DD / BNT), 256, 0, stream>>>(Ob, wob, out, M, DD, DD);
}
